// GraphTransformerLayer_78864189489249
// MI455X (gfx1250) — compile-verified, matrix operand order checked by reading, not run on hardware
//
#include <hip/hip_runtime.h>

// ---------------------------------------------------------------------------
// GraphTransformerLayer for MI455X (gfx1250, wave32, WMMA bf16 + TDM path)
// N=4096 nodes, D=512, H=8 heads, DH=64, E=131072 edges
// ---------------------------------------------------------------------------

typedef unsigned int       u32;
typedef unsigned short     u16;
typedef unsigned long long u64;
typedef long long          i64;

typedef __attribute__((ext_vector_type(16))) __bf16       v16bf;
typedef __attribute__((ext_vector_type(8)))  float        v8f;
typedef __attribute__((ext_vector_type(4)))  unsigned int v4u;
typedef __attribute__((ext_vector_type(8)))  int          v8i;
typedef __attribute__((ext_vector_type(4)))  int          v4i;

union FragAB { v16bf v; u32 w[8]; u16 h[16]; };

static constexpr int N_  = 4096;
static constexpr int D_  = 512;
static constexpr int H_  = 8;
static constexpr int DH_ = 64;
static constexpr int E_  = 131072;
static constexpr int D3_ = 3 * D_;   // 1536
static constexpr int D4_ = 4 * D_;   // 2048
static constexpr int MASKW_ = N_ / 32;  // 128 words per row

#if __has_builtin(__builtin_amdgcn_tensor_load_to_lds)
#define USE_TDM 1
#else
#define USE_TDM 0
#endif

// float -> bf16 bits, round-to-nearest-even
__device__ __forceinline__ u16 f2bf(float f) {
  u32 u = __builtin_bit_cast(u32, f);
  u32 r = u + 0x7FFFu + ((u >> 16) & 1u);
  return (u16)(r >> 16);
}

// 16-bit A-matrix 16x32 VGPR layout: pair base K for u32 slot i of this lane.
// lanes 0-15: K = {0,2,4,6,16,18,20,22}; lanes 16-31: +8
__device__ __forceinline__ int kpair(int i, int lane) {
  int kb = (lane < 16) ? 0 : 8;
  return ((i < 4) ? (2 * i) : (16 + 2 * (i - 4))) + kb;
}

#if USE_TDM
// ---------------------------------------------------------------------------
// TDM: issue a 2D tensor_load_to_lds of a 64-row x 32-col bf16 tile.
// Rows are k_stride elements apart in memory; tile lands row-major [64][32]
// contiguously at lds_off. D# packing per CDNA5 ISA ch.8 (group0/group1).
// ---------------------------------------------------------------------------
__device__ __forceinline__ void tdm_load_tile(u32 lds_off, const u16* gptr, u32 k_stride) {
  u64 ga = (u64)(size_t)gptr;
  v4u g0 = { 1u,                       // count=1 (valid user descriptor)
             lds_off,                  // LDS byte address
             (u32)ga,                  // global addr [31:0]
             (u32)(ga >> 32) | (2u << 30) };  // addr[56:32] | type=2
  v8i g1 = {
    (int)(1u << 16),                               // data_size=1 -> 2 bytes
    (int)((k_stride & 0xFFFFu) << 16),             // tensor_dim0[15:0]
    (int)((k_stride >> 16) | (0xFFFFu << 16)),     // dim0[31:16] | dim1[15:0]
    (int)(0u | (32u << 16)),                       // dim1[31:16] | tile_dim0=32
    (int)(64u),                                    // tile_dim1=64, tile_dim2=0
    (int)k_stride,                                 // tensor_dim0_stride[31:0]
    0,                                             // stride0[47:32] | stride1 lo
    0 };
  v4i g2 = {0, 0, 0, 0};
  v4i g3 = {0, 0, 0, 0};
#if defined(__clang_major__) && __clang_major__ >= 23
  v8i g4 = {0, 0, 0, 0, 0, 0, 0, 0};
  __builtin_amdgcn_tensor_load_to_lds(g0, g1, g2, g3, g4, 0);
#else
  __builtin_amdgcn_tensor_load_to_lds(g0, g1, g2, g3, 0);
#endif
}
#endif

// ---------------------------------------------------------------------------
// weight f32 -> bf16 conversion
// ---------------------------------------------------------------------------
__global__ void k_convert_bf16(const float* __restrict__ in, u16* __restrict__ out, int n) {
  int i = blockIdx.x * 256 + threadIdx.x;
  if (i < n) out[i] = f2bf(in[i]);
}

// ---------------------------------------------------------------------------
// adjacency bit-mask: adj[r,c]=1 for edges + diagonal
// ---------------------------------------------------------------------------
__global__ void k_build_mask(const i64* __restrict__ ei, u32* __restrict__ mask) {
  int i = blockIdx.x * 256 + threadIdx.x;
  if (i < E_) {
    int r = (int)ei[i];
    int c = (int)ei[E_ + i];
    atomicOr(&mask[r * MASKW_ + (c >> 5)], 1u << (c & 31));
  } else if (i < E_ + N_) {
    int d = i - E_;
    atomicOr(&mask[d * MASKW_ + (d >> 5)], 1u << (d & 31));
  }
}

// ---------------------------------------------------------------------------
// LayerNorm: one wave per row, 16 elems per lane, shfl_xor reductions, bf16 out
// ---------------------------------------------------------------------------
__global__ __launch_bounds__(256) void k_layernorm(const float* __restrict__ x,
                                                   const float* __restrict__ w,
                                                   const float* __restrict__ b,
                                                   u16* __restrict__ out) {
  int wave = threadIdx.x >> 5;
  int lane = threadIdx.x & 31;
  int row  = blockIdx.x * 8 + wave;
  const float* xr = x + (size_t)row * D_;
  float s = 0.f, ss = 0.f;
#pragma unroll
  for (int i = 0; i < D_ / 32; i++) {
    float v = xr[lane + 32 * i];
    s += v; ss += v * v;
  }
#pragma unroll
  for (int off = 16; off; off >>= 1) {
    s  += __shfl_xor(s,  off, 32);
    ss += __shfl_xor(ss, off, 32);
  }
  float mean = s * (1.0f / D_);
  float var  = ss * (1.0f / D_) - mean * mean;
  float rstd = rsqrtf(var + 1e-5f);
#pragma unroll
  for (int i = 0; i < D_ / 32; i++) {
    int c = lane + 32 * i;
    float v = (xr[c] - mean) * rstd * w[c] + b[c];
    out[(size_t)row * D_ + c] = f2bf(v);
  }
}

// ---------------------------------------------------------------------------
// bf16 GEMM: C[M x J] = A[M x K] * B[J x K]^T + bias.
// TDM double-buffered LDS staging (wave 0 drives the Tensor Data Mover),
// WMMA bf16 compute. block = 128 threads (4 waves), tile 64x64, K-step 32.
// MODE 0: store bf16 (+ optional transposed store of cols >= 1024 into vt)
// MODE 1: relu + store bf16. MODE 2: f32 out = C + res.
// ---------------------------------------------------------------------------
template <int MODE>
__global__ __launch_bounds__(128) void k_gemm_bf16(const u16* __restrict__ A,
                                                   const u16* __restrict__ B,
                                                   const float* __restrict__ bias,
                                                   const float* __restrict__ res,
                                                   void* __restrict__ outp,
                                                   u16* __restrict__ vt,
                                                   int K, int J) {
  __shared__ __align__(16) u16 As[2][64 * 32];
  __shared__ __align__(16) u16 Bs[2][64 * 32];
  const int tid  = threadIdx.x;
  const int lane = tid & 31;
  const int wv   = tid >> 5;
  const int mblk = blockIdx.x * 64;
  const int jblk = blockIdx.y * 64;
  const int m    = lane & 15;
  const int nK   = K / 32;

  v8f acc[4];
#pragma unroll
  for (int t = 0; t < 4; t++) acc[t] = (v8f){0.f, 0.f, 0.f, 0.f, 0.f, 0.f, 0.f, 0.f};

  const u16* Abase = A + (size_t)mblk * K;
  const u16* Bbase = B + (size_t)jblk * K;

#if USE_TDM
  if (wv == 0) {
    tdm_load_tile((u32)(size_t)&As[0][0], Abase, (u32)K);
    tdm_load_tile((u32)(size_t)&Bs[0][0], Bbase, (u32)K);
  }
#else
  const int lr = tid >> 1;
  const int lc = (tid & 1) * 16;
#endif

  for (int kb = 0; kb < nK; kb++) {
    const int cur = kb & 1;
#if USE_TDM
    if (wv == 0) {
      if (kb + 1 < nK) {
        tdm_load_tile((u32)(size_t)&As[cur ^ 1][0], Abase + (size_t)(kb + 1) * 32, (u32)K);
        tdm_load_tile((u32)(size_t)&Bs[cur ^ 1][0], Bbase + (size_t)(kb + 1) * 32, (u32)K);
        __builtin_amdgcn_s_wait_tensorcnt(2);  // current pair complete
      } else {
        __builtin_amdgcn_s_wait_tensorcnt(0);
      }
    }
    __syncthreads();
#else
    {
      const uint4* ga = reinterpret_cast<const uint4*>(Abase + (size_t)lr * K + kb * 32 + lc);
      const uint4* gb = reinterpret_cast<const uint4*>(Bbase + (size_t)lr * K + kb * 32 + lc);
      uint4 a0 = ga[0], a1 = ga[1];
      uint4 b0 = gb[0], b1 = gb[1];
      __syncthreads();
      *reinterpret_cast<uint4*>(&As[cur][lr * 32 + lc])     = a0;
      *reinterpret_cast<uint4*>(&As[cur][lr * 32 + lc + 8]) = a1;
      *reinterpret_cast<uint4*>(&Bs[cur][lr * 32 + lc])     = b0;
      *reinterpret_cast<uint4*>(&Bs[cur][lr * 32 + lc + 8]) = b1;
      __syncthreads();
    }
#endif

    FragAB af;
#pragma unroll
    for (int i = 0; i < 8; i++) {
      int kp = kpair(i, lane);
      af.w[i] = *reinterpret_cast<const u32*>(&As[cur][(wv * 16 + m) * 32 + kp]);
    }
#pragma unroll
    for (int nt = 0; nt < 4; nt++) {
      FragAB bfr;
#pragma unroll
      for (int i = 0; i < 8; i++) {
        int kp = kpair(i, lane);
        bfr.w[i] = *reinterpret_cast<const u32*>(&Bs[cur][(nt * 16 + m) * 32 + kp]);
      }
      acc[nt] = __builtin_amdgcn_wmma_f32_16x16x32_bf16(
          false, af.v, false, bfr.v, (short)0, acc[nt], false, false);
    }
    __syncthreads();  // all waves done with buf before it is refilled
  }

  // epilogue: C/D layout -> lane holds col n, rows r (+8 for upper half)
  const int n      = lane & 15;
  const int rowoff = (lane < 16) ? 0 : 8;
#pragma unroll
  for (int nt = 0; nt < 4; nt++) {
    int j = jblk + nt * 16 + n;
    float bb = bias[j];
#pragma unroll
    for (int rr = 0; rr < 8; rr++) {
      int grow = mblk + wv * 16 + rowoff + rr;
      float val = acc[nt][rr] + bb;
      if constexpr (MODE == 0) {
        if (vt != nullptr && j >= 2 * D_) {
          // V goes out transposed: vt[dh_global][node] for contiguous
          // key-pairs in the attention P*V B-fragment
          vt[(size_t)(j - 2 * D_) * N_ + grow] = f2bf(val);
        } else {
          ((u16*)outp)[(size_t)grow * J + j] = f2bf(val);
        }
      } else if constexpr (MODE == 1) {
        ((u16*)outp)[(size_t)grow * J + j] = f2bf(fmaxf(val, 0.f));
      } else {
        ((float*)outp)[(size_t)grow * J + j] = val + res[(size_t)grow * J + j];
      }
    }
  }
}

// ---------------------------------------------------------------------------
// Flash-style masked multi-head attention. One wave per (head, 16 q-rows).
// qkv: bf16 [N][1536] (q|k valid), vt: bf16 [D][N] (v transposed).
// Per 32-key block: S = Q K^T (4 wmma), online softmax, O += P V (4 wmma).
// ---------------------------------------------------------------------------
__global__ __launch_bounds__(32) void k_attention(const u16* __restrict__ qkv,
                                                  const u16* __restrict__ vt,
                                                  const u32* __restrict__ mask,
                                                  u16* __restrict__ o) {
  __shared__ __align__(4) u16 Plds[16 * 32];
  const int lane = threadIdx.x;
  const int h    = blockIdx.x & (H_ - 1);
  const int qb   = blockIdx.x >> 3;
  const int m    = lane & 15;
  const int rowoff = (lane < 16) ? 0 : 8;

  // Q fragments (16 rows x 64 d, split into two K=32 fragments)
  FragAB qf[2];
#pragma unroll
  for (int dblk = 0; dblk < 2; dblk++) {
#pragma unroll
    for (int i = 0; i < 8; i++) {
      int kp = kpair(i, lane);
      qf[dblk].w[i] = *reinterpret_cast<const u32*>(
          &qkv[(size_t)(qb * 16 + m) * D3_ + h * DH_ + dblk * 32 + kp]);
    }
  }

  v8f Oacc[4];
#pragma unroll
  for (int t = 0; t < 4; t++) Oacc[t] = (v8f){0.f, 0.f, 0.f, 0.f, 0.f, 0.f, 0.f, 0.f};
  float run_m[8], run_l[8];
#pragma unroll
  for (int r = 0; r < 8; r++) { run_m[r] = -3.0e38f; run_l[r] = 0.f; }

  const float scale = 0.125f;  // 1/sqrt(DH)

  for (int kb = 0; kb < N_ / 32; kb++) {
    const int key0 = kb * 32;

    // ---- S = Q K^T : two 16x16 tiles covering 32 keys ----
    v8f s0 = (v8f){0.f,0.f,0.f,0.f,0.f,0.f,0.f,0.f};
    v8f s1 = (v8f){0.f,0.f,0.f,0.f,0.f,0.f,0.f,0.f};
#pragma unroll
    for (int dblk = 0; dblk < 2; dblk++) {
      FragAB k0f, k1f;
#pragma unroll
      for (int i = 0; i < 8; i++) {
        int kp = kpair(i, lane);
        k0f.w[i] = *reinterpret_cast<const u32*>(
            &qkv[(size_t)(key0 + m) * D3_ + D_ + h * DH_ + dblk * 32 + kp]);
        k1f.w[i] = *reinterpret_cast<const u32*>(
            &qkv[(size_t)(key0 + 16 + m) * D3_ + D_ + h * DH_ + dblk * 32 + kp]);
      }
      s0 = __builtin_amdgcn_wmma_f32_16x16x32_bf16(false, qf[dblk].v, false, k0f.v,
                                                   (short)0, s0, false, false);
      s1 = __builtin_amdgcn_wmma_f32_16x16x32_bf16(false, qf[dblk].v, false, k1f.v,
                                                   (short)0, s1, false, false);
    }

    // ---- mask + online softmax ----
    u32 mw = 0;
    if (lane < 16) mw = mask[(size_t)(qb * 16 + lane) * MASKW_ + kb];
    float p0[8], p1[8];
#pragma unroll
    for (int rr = 0; rr < 8; rr++) {
      int mrow = rowoff + rr;
      u32 wbits = (u32)__shfl((int)mw, mrow, 32);
      float a0 = s0[rr] * scale + (((wbits >> m) & 1u) ? 0.f : -1.0e9f);
      float a1 = s1[rr] * scale + (((wbits >> (16 + m)) & 1u) ? 0.f : -1.0e9f);
      float t = fmaxf(a0, a1);
#pragma unroll
      for (int off = 1; off < 16; off <<= 1) t = fmaxf(t, __shfl_xor(t, off, 32));
      float nm    = fmaxf(run_m[rr], t);
      float alpha = __expf(run_m[rr] - nm);
      float e0 = __expf(a0 - nm);
      float e1 = __expf(a1 - nm);
      float rs = e0 + e1;
#pragma unroll
      for (int off = 1; off < 16; off <<= 1) rs += __shfl_xor(rs, off, 32);
      run_l[rr] = run_l[rr] * alpha + rs;
      run_m[rr] = nm;
      p0[rr] = e0; p1[rr] = e1;
#pragma unroll
      for (int nt = 0; nt < 4; nt++) Oacc[nt][rr] *= alpha;
    }

    // ---- transpose P through LDS into A-fragment layout ----
#pragma unroll
    for (int rr = 0; rr < 8; rr++) {
      int mrow = rowoff + rr;
      Plds[mrow * 32 + m]      = f2bf(p0[rr]);
      Plds[mrow * 32 + 16 + m] = f2bf(p1[rr]);
    }
    __syncthreads();
    FragAB pf;
#pragma unroll
    for (int i = 0; i < 8; i++) {
      int kp = kpair(i, lane);
      pf.w[i] = *reinterpret_cast<const u32*>(&Plds[m * 32 + kp]);
    }

    // ---- O += P V : 4 tiles over DH, V read from transposed vt ----
#pragma unroll
    for (int nt = 0; nt < 4; nt++) {
      FragAB vf;
#pragma unroll
      for (int i = 0; i < 8; i++) {
        int kp = kpair(i, lane);
        vf.w[i] = *reinterpret_cast<const u32*>(
            &vt[(size_t)(h * DH_ + nt * 16 + m) * N_ + key0 + kp]);
      }
      Oacc[nt] = __builtin_amdgcn_wmma_f32_16x16x32_bf16(
          false, pf.v, false, vf.v, (short)0, Oacc[nt], false, false);
    }
    __syncthreads();
  }

  // ---- normalize + store O (bf16, [N][D], head-major columns) ----
#pragma unroll
  for (int rr = 0; rr < 8; rr++) {
    float inv = __builtin_amdgcn_rcpf(run_l[rr]);
    int mrow = rowoff + rr;
#pragma unroll
    for (int nt = 0; nt < 4; nt++) {
      float val = Oacc[nt][rr] * inv;
      o[(size_t)(qb * 16 + mrow) * D_ + h * DH_ + nt * 16 + m] = f2bf(val);
    }
  }
}

// ---------------------------------------------------------------------------
// host-side orchestration
// ---------------------------------------------------------------------------
extern "C" void kernel_launch(void* const* d_in, const int* in_sizes, int n_in,
                              void* d_out, int out_size, void* d_ws, size_t ws_size,
                              hipStream_t stream) {
  (void)in_sizes; (void)n_in; (void)out_size; (void)ws_size;

  const float* x     = (const float*)d_in[0];
  const i64*   ei    = (const i64*)  d_in[1];
  const float* ln1_w = (const float*)d_in[2];
  const float* ln1_b = (const float*)d_in[3];
  const float* w_in  = (const float*)d_in[4];
  const float* b_in  = (const float*)d_in[5];
  const float* w_out = (const float*)d_in[6];
  const float* b_out = (const float*)d_in[7];
  const float* ln2_w = (const float*)d_in[8];
  const float* ln2_b = (const float*)d_in[9];
  const float* w1    = (const float*)d_in[10];
  const float* b1    = (const float*)d_in[11];
  const float* w2    = (const float*)d_in[12];
  const float* b2    = (const float*)d_in[13];
  float* out = (float*)d_out;

  unsigned char* ws = (unsigned char*)d_ws;
  // workspace layout (bytes)
  u32* mask    = (u32*)(ws + 0);                 //  2 MB  : N x 128 words
  u16* w_in_b  = (u16*)(ws + 2097152);           //  1.5 MB: 1536x512 bf16
  u16* w_out_b = (u16*)(ws + 3670016);           //  0.5 MB: 512x512
  u16* w1_b    = (u16*)(ws + 4194304);           //  2 MB  : 2048x512
  u16* w2_b    = (u16*)(ws + 6291456);           //  2 MB  : 512x2048
  u16* xn      = (u16*)(ws + 8388608);           //  4 MB  : N x 512
  u16* qkv     = (u16*)(ws + 12582912);          // 12 MB  : N x 1536 (q|k)
  u16* o_bf    = (u16*)(ws + 25165824);          //  4 MB  : N x 512
  float* x2    = (float*)(ws + 29360128);        //  8 MB  : N x 512 f32
  u16* xm      = (u16*)(ws + 37748736);          //  4 MB  : N x 512
  u16* hid     = (u16*)(ws + 41943040);          // 16 MB  : N x 2048
  u16* vt      = (u16*)(ws + 58720256);          //  4 MB  : D x N (v^T)

  // 1) adjacency mask
  hipMemsetAsync(mask, 0, (size_t)N_ * MASKW_ * 4, stream);
  k_build_mask<<<(E_ + N_ + 255) / 256, 256, 0, stream>>>(ei, mask);

  // 2) weights -> bf16
  k_convert_bf16<<<(D3_ * D_ + 255) / 256, 256, 0, stream>>>(w_in,  w_in_b,  D3_ * D_);
  k_convert_bf16<<<(D_  * D_ + 255) / 256, 256, 0, stream>>>(w_out, w_out_b, D_  * D_);
  k_convert_bf16<<<(D4_ * D_ + 255) / 256, 256, 0, stream>>>(w1,    w1_b,    D4_ * D_);
  k_convert_bf16<<<(D_ * D4_ + 255) / 256, 256, 0, stream>>>(w2,    w2_b,    D_ * D4_);

  // 3) LN1
  k_layernorm<<<N_ / 8, 256, 0, stream>>>(x, ln1_w, ln1_b, xn);

  // 4) QKV projection: [N,512] x [1536,512]^T -> q|k in qkv, v transposed in vt
  k_gemm_bf16<0><<<dim3(N_ / 64, D3_ / 64), 128, 0, stream>>>(
      xn, w_in_b, b_in, nullptr, qkv, vt, D_, D3_);

  // 5) masked attention -> bf16 [N,512]
  k_attention<<<(N_ / 16) * H_, 32, 0, stream>>>(qkv, vt, mask, o_bf);

  // 6) out projection + residual: x2 = x + o w_out^T + b_out  (f32)
  k_gemm_bf16<2><<<dim3(N_ / 64, D_ / 64), 128, 0, stream>>>(
      o_bf, w_out_b, b_out, x, x2, nullptr, D_, D_);

  // 7) LN2
  k_layernorm<<<N_ / 8, 256, 0, stream>>>(x2, ln2_w, ln2_b, xm);

  // 8) MLP up + relu: bf16 [N,2048]
  k_gemm_bf16<1><<<dim3(N_ / 64, D4_ / 64), 128, 0, stream>>>(
      xm, w1_b, b1, nullptr, hid, nullptr, D_, D4_);

  // 9) MLP down + residual -> d_out (f32)
  k_gemm_bf16<2><<<dim3(N_ / 64, D_ / 64), 128, 0, stream>>>(
      hid, w2_b, b2, x2, out, nullptr, D4_, D_);
}
